// StyledLayer_68805376082027
// MI455X (gfx1250) — compile-verified, matrix operand order checked by reading, not run on hardware
//
#include <hip/hip_runtime.h>
#include <hip/hip_bf16.h>

// ---------------------------------------------------------------------------
// StyleGAN2 modulated conv, refactored:
//   y[b,ci]      = (w @ style_weight.T)/sqrt(SDIM) + bias
//   xm[b,ci,..]  = y[b,ci] * x[b,ci,..]            (folded into patch staging)
//   out[b,co,..] = demod[b,co] * conv2d(xm, conv_scale*W)  (+noise, lrelu*sqrt2)
//   demod[b,co]  = rsqrt(conv_scale^2 * sum_ci y^2 * wsq[co,ci] + 1e-8)
// Implicit GEMM: M = 4096 pixels/sample, N = 512, K = 512*9, bf16 WMMA f32-acc.
// B tiles (all 9 taps) DMA'd into LDS by the Tensor Data Mover each ci-step,
// overlapped with VALU staging of the modulated A halo patch.
// ---------------------------------------------------------------------------

#define NB    16
#define CIN   512
#define COUT  512
#define SDIM  512
#define HWPX  4096

typedef __attribute__((ext_vector_type(16))) __bf16       v16bf;
typedef __attribute__((ext_vector_type(8)))  float        v8f;
typedef __attribute__((ext_vector_type(4)))  unsigned int u32x4;
typedef __attribute__((ext_vector_type(8)))  int          i32x8;
typedef __attribute__((ext_vector_type(4)))  int          i32x4;

union FragBF { u32x4 u[2]; v16bf v; };

#if defined(__has_builtin)
#  if __has_builtin(__builtin_amdgcn_tensor_load_to_lds)
#    define HAVE_TDM 1
#  endif
#endif
#ifndef HAVE_TDM
#  define HAVE_TDM 0
#endif

__device__ __forceinline__ unsigned short f2bf(float f) {
    unsigned int u = __builtin_bit_cast(unsigned int, f);
    u += 0x7fffu + ((u >> 16) & 1u);          // round-to-nearest-even
    return (unsigned short)(u >> 16);
}

// y[b,ci] = dot(w[b,:], style_weight[ci,:]) / sqrt(SDIM) + style_bias[ci]
__global__ void style_y_kernel(const float* __restrict__ w,
                               const float* __restrict__ sw,
                               const float* __restrict__ sb,
                               float* __restrict__ y) {
    int id = blockIdx.x * 256 + threadIdx.x;      // NB*CIN = 8192
    int b  = id >> 9;
    int ci = id & 511;
    const float* wr = w  + b  * SDIM;
    const float* sr = sw + ci * SDIM;
    float acc = 0.f;
    for (int s = 0; s < SDIM; ++s) acc += wr[s] * sr[s];
    y[id] = acc * 0.04419417382415922f /* 1/sqrt(512) */ + sb[ci];
}

// wsq[co,ci] = sum_k W^2 ; wb[tap][co][ci] = bf16(conv_scale * W)
__global__ void prep_w_kernel(const float* __restrict__ cw,
                              float* __restrict__ wsq,
                              unsigned short* __restrict__ wb) {
    int id = blockIdx.x * 256 + threadIdx.x;      // COUT*CIN = 262144
    int co = id >> 9;
    int ci = id & 511;
    const float cscale = 0.014731391274719739f;   // 1/sqrt(512*9)
    const float* src = cw + id * 9;
    float s = 0.f;
#pragma unroll
    for (int t = 0; t < 9; ++t) {
        float v = src[t];
        s += v * v;
        wb[(t * COUT + co) * CIN + ci] = f2bf(v * cscale);
    }
    wsq[id] = s;
}

// demod[b,co] = rsqrt(conv_scale^2 * sum_ci y^2 * wsq + 1e-8)
__global__ void demod_kernel(const float* __restrict__ y,
                             const float* __restrict__ wsq,
                             float* __restrict__ dm) {
    int id = blockIdx.x * 256 + threadIdx.x;      // NB*COUT = 8192
    int b  = id >> 9;
    int co = id & 511;
    const float* yr = y   + b  * CIN;
    const float* wr = wsq + co * CIN;
    float acc = 0.f;
    for (int ci = 0; ci < CIN; ++ci) {
        float yv = yr[ci];
        acc += yv * yv * wr[ci];
    }
    dm[id] = rsqrtf(2.170138888888889e-4f /* 1/4608 */ * acc + 1e-8f);
}

// LDS layout (dynamic, 122496 B -> 2 blocks per 320KB WGP):
//   Apatch : 4 rows x 66 cols x 32 ci bf16, pixel stride 40 ushorts (80 B):
//            tap shifts move whole pixels so b128 fragment reads stay 16B
//            aligned; 20-dword stride => conflict-free fragment rows.
//   Bsh    : 9 taps x 128 couts x 32 ci bf16, row stride 40 ushorts.
//            Filled by TDM: 16-dword rows + 4-dword pad (pad_interval=3,
//            pad_amount=3) reproduces the 80-B stride in hardware.
//   Tsh    : per-wave 16x18 f32 transpose buffer for coalesced stores.
#define APATCH_OFF 0
#define BSH_OFF    21120
#define TSH_OFF    (21120 + 92160)
#define SMEM_BYTES (21120 + 92160 + 9216)

__global__ __launch_bounds__(256) void conv_wmma_kernel(
        const float* __restrict__ x, const unsigned short* __restrict__ wb,
        const float* __restrict__ y, const float* __restrict__ dm,
        const float* __restrict__ noise, const float* __restrict__ nw,
        float* __restrict__ out) {
    extern __shared__ unsigned char smem[];
    unsigned short* Apatch = (unsigned short*)(smem + APATCH_OFF);
    unsigned short* Bsh    = (unsigned short*)(smem + BSH_OFF);
    float*          Tsh    = (float*)(smem + TSH_OFF);

    const int b    = blockIdx.z;
    const int p0   = blockIdx.x * 128;            // pixel tile (2 image rows)
    const int h0   = p0 >> 6;
    const int n0   = blockIdx.y * 128;            // cout tile base
    const int t    = threadIdx.x;
    const int lane = t & 31;
    const int wave = t >> 5;
    const int wm   = wave & 3;                    // 4 waves along M
    const int wn   = wave >> 2;                   // 2 waves along N
    const int lr   = lane & 15;
    const int kb   = (lane >> 4) << 3;            // 0 | 8 (ISA 16-bit A layout)

    v8f acc[2][4] = {};

    const float* xb = x + (size_t)b * CIN * HWPX;
    const float* yb = y + b * CIN;

    const int ciL = t >> 3;                       // 0..31 : ci lane this thread fills
    const int sl  = t & 7;                        // 0..7  : slot sub-index

    for (int k0 = 0; k0 < CIN; k0 += 32) {
        // ---- kick off TDM: one 3D tile = 9 taps x 128 co x 32 ci bf16 ----
        // (previous-iteration __syncthreads guarantees Bsh is free)
#if HAVE_TDM
        if (wave == 0) {
            unsigned long long ga =
                (unsigned long long)(const void*)(wb + (size_t)n0 * CIN + k0);
            unsigned int ldsa = (unsigned int)(unsigned long long)(void*)Bsh;
            u32x4 g0 = { 1u,                                  // count=1 (user D#)
                         ldsa,                                // lds_addr
                         (unsigned int)ga,                    // global_addr lo
                         (unsigned int)((ga >> 32) & 0x01FFFFFFull) | (2u << 30) };
            i32x8 g1 = { (1 << 16) | (1 << 20) | (3 << 22) | (3 << 25),
                         //  ^data_size=2B  ^pad_en  ^interval=16dw  ^amount=4dw
                         (int)(512u << 16),                   // tensor_dim0 = 512
                         (int)(4608u << 16),                  // tensor_dim1 = 4608
                         (32 << 16),                          // tile_dim0 = 32
                         128 | (9 << 16),                     // tile_dim1, tile_dim2
                         512,                                 // dim0_stride = 512
                         0,
                         4 };                                 // dim1_stride = 512*512
            i32x4 g2 = { 9, 0, 0, 0 };                        // tensor_dim2 = 9
            i32x4 g3 = { 0, 0, 0, 0 };
#if __clang_major__ >= 23
            i32x8 gz = { 0, 0, 0, 0, 0, 0, 0, 0 };
            __builtin_amdgcn_tensor_load_to_lds(g0, g1, g2, g3, gz, 0);
#else
            __builtin_amdgcn_tensor_load_to_lds(g0, g1, g2, g3, 0);
#endif
        }
#endif
        // ---- stage halo patch: 4 rows x 66 cols x 32 ci, modulate+cvt ----
        // (runs concurrently with the TDM transfer)
        {
            const int    ci   = k0 + ciL;
            const float  ym   = yb[ci];
            const float* xrow = xb + (size_t)ci * HWPX;
#pragma unroll
            for (int j = 0; j < 33; ++j) {
                int s = j * 8 + sl;               // 264 slots total
                if (s < 264) {
                    int   prow = s / 66;
                    int   pcol = s - prow * 66;
                    int   hh   = h0 - 1 + prow;
                    int   ww   = pcol - 1;
                    float v    = 0.f;
                    if ((unsigned)hh < 64u && (unsigned)ww < 64u)
                        v = xrow[hh * 64 + ww];
                    Apatch[s * 40 + ciL] = f2bf(v * ym);
                }
            }
            if (k0 + 32 < CIN)                    // prefetch next ci-block of x
                __builtin_prefetch(xrow + 32 * HWPX + (size_t)h0 * 64, 0, 1);
        }
#if HAVE_TDM
        if (wave == 0)
            __builtin_amdgcn_s_wait_tensorcnt(0);
#else
        {   // fallback: manual B staging
#pragma unroll
            for (int i = 0; i < 9; ++i) {
                int idx  = i * 256 + t;           // 2304 half-rows
                int half = idx & 1;
                int row  = idx >> 1;              // tap*128 + co
                int tap  = row >> 7;
                int co   = row & 127;
                const u32x4* src = (const u32x4*)(
                    wb + ((size_t)tap * COUT + n0 + co) * CIN + k0 + half * 16);
                u32x4 v0 = src[0];
                u32x4 v1 = src[1];
                *(u32x4*)&Bsh[row * 40 + half * 16]     = v0;
                *(u32x4*)&Bsh[row * 40 + half * 16 + 8] = v1;
            }
        }
#endif
        __syncthreads();

        // ---- 9 taps x (2x4) WMMAs, A re-read from shifted patch ----------
#pragma unroll
        for (int kh = -1; kh <= 1; ++kh) {
#pragma unroll
            for (int kw = -1; kw <= 1; ++kw) {
                const int tap = (kh + 1) * 3 + (kw + 1);
                FragBF a[2], bfr[4];
#pragma unroll
                for (int sm = 0; sm < 2; ++sm) {
                    int m    = wm * 32 + sm * 16 + lr;
                    int base = (((m >> 6) + kh + 1) * 66 + (m & 63) + kw + 1) * 40;
                    a[sm].u[0] = *(const u32x4*)&Apatch[base + kb];
                    a[sm].u[1] = *(const u32x4*)&Apatch[base + 16 + kb];
                }
#pragma unroll
                for (int sn = 0; sn < 4; ++sn) {
                    int col   = wn * 64 + sn * 16 + lr;
                    int bbase = (tap * 128 + col) * 40;
                    bfr[sn].u[0] = *(const u32x4*)&Bsh[bbase + kb];
                    bfr[sn].u[1] = *(const u32x4*)&Bsh[bbase + 16 + kb];
                }
#pragma unroll
                for (int sm = 0; sm < 2; ++sm)
#pragma unroll
                    for (int sn = 0; sn < 4; ++sn)
                        acc[sm][sn] = __builtin_amdgcn_wmma_f32_16x16x32_bf16(
                            false, a[sm].v, false, bfr[sn].v,
                            (short)0, acc[sm][sn], false, false);
            }
        }
        __syncthreads();
    }

    // ---- epilogue: demod, noise, lrelu*sqrt2; LDS transpose for coalesced
    //      global_store_b128 (8 contiguous pixels per lane per store pair) --
    const float  ns   = nw[0];
    const float* nrow = noise + (size_t)b * HWPX;
    const float* drow = dm + b * COUT;
    const int    mhi  = (lane >> 4) << 3;         // lanes 16-31 hold M+8
    float*       tw   = Tsh + wave * (16 * 18);
    const int    cr   = lane >> 1;                // co row within 16x16 tile
    const int    ph   = (lane & 1) * 8;           // pixel sub-offset
#pragma unroll
    for (int sm = 0; sm < 2; ++sm) {
        const int pbase = p0 + wm * 32 + sm * 16;
#pragma unroll
        for (int sn = 0; sn < 4; ++sn) {
            const int   cob = n0 + wn * 64 + sn * 16;
            const float d   = drow[cob + lr];
#pragma unroll
            for (int r = 0; r < 8; ++r) {
                int   m = r + mhi;
                float v = acc[sm][sn][r] * d + ns * nrow[pbase + m];
                v = (v < 0.f ? 0.2f * v : v) * 1.4142135623730951f;
                tw[m * 18 + lr] = v;
            }
            __syncthreads();
            float4 f0, f1;
            f0.x = tw[(ph + 0) * 18 + cr];
            f0.y = tw[(ph + 1) * 18 + cr];
            f0.z = tw[(ph + 2) * 18 + cr];
            f0.w = tw[(ph + 3) * 18 + cr];
            f1.x = tw[(ph + 4) * 18 + cr];
            f1.y = tw[(ph + 5) * 18 + cr];
            f1.z = tw[(ph + 6) * 18 + cr];
            f1.w = tw[(ph + 7) * 18 + cr];
            float* orow = out + ((size_t)b * COUT + cob + cr) * HWPX + pbase + ph;
            *(float4*)(orow)     = f0;
            *(float4*)(orow + 4) = f1;
            __syncthreads();
        }
    }
}

extern "C" void kernel_launch(void* const* d_in, const int* in_sizes, int n_in,
                              void* d_out, int out_size, void* d_ws, size_t ws_size,
                              hipStream_t stream) {
    const float* x     = (const float*)d_in[0];   // [16,512,64,64]
    const float* w     = (const float*)d_in[1];   // [16,512]
    const float* noise = (const float*)d_in[2];   // [16,1,64,64]
    const float* sw    = (const float*)d_in[3];   // [512,512]
    const float* sb    = (const float*)d_in[4];   // [512]
    const float* cw    = (const float*)d_in[5];   // [512,512,3,3]
    const float* nw    = (const float*)d_in[6];   // [1]
    float*       out   = (float*)d_out;           // [16,512,64,64]

    char* ws = (char*)d_ws;
    float*          y   = (float*)(ws);                            // 32 KB
    float*          dmv = (float*)(ws + 32768);                    // 32 KB
    float*          wsq = (float*)(ws + 65536);                    // 1 MB
    unsigned short* wb  = (unsigned short*)(ws + 65536 + 1048576); // 4.5 MB bf16

    style_y_kernel<<<32, 256, 0, stream>>>(w, sw, sb, y);
    prep_w_kernel<<<1024, 256, 0, stream>>>(cw, wsq, wb);
    demod_kernel<<<32, 256, 0, stream>>>(y, wsq, dmv);

    dim3 grid(HWPX / 128, COUT / 128, NB);        // (32, 4, 16)
    conv_wmma_kernel<<<grid, 256, SMEM_BYTES, stream>>>(x, wb, y, dmv, noise, nw, out);
}
